// NoisyTopKGate_77051713290692
// MI455X (gfx1250) — matrix operand order repaired and checked by reading, not compiled, not run on hardware
//
#include <hip/hip_runtime.h>
#include <cmath>

// ---- problem constants (from reference) ----
#define DIM     2048
#define NEXP    64
#define NCOLS   128      // 64 Wg cols + 64 Wn cols fused
#define TOPK    8
#define NBATCH  16384
#define KB      64       // K per main-loop iteration
#define MB      64       // rows per block
#define LDSPAD  8        // ushort pad -> keeps 16B alignment, spreads banks

typedef __attribute__((ext_vector_type(16))) __bf16        v16bf;
typedef __attribute__((ext_vector_type(8)))  float         v8f;
typedef __attribute__((ext_vector_type(4)))  unsigned int  u32x4;

union Frag { v16bf v; u32x4 q[2]; };

__device__ __forceinline__ unsigned short f32_to_bf16(float f) {
  unsigned int u = __builtin_bit_cast(unsigned int, f);
  u += 0x7FFFu + ((u >> 16) & 1u);           // round-to-nearest-even
  return (unsigned short)(u >> 16);
}

__device__ __forceinline__ float softplus_stable(float v) {
  return fmaxf(v, 0.0f) + log1pf(__expf(-fabsf(v)));
}

// ---------------------------------------------------------------------------
// Kernel 1: pack Wg (64x2048) and Wn (64x2048) into one bf16 [128][2048]
// ---------------------------------------------------------------------------
__global__ __launch_bounds__(256) void moe_convert_weights(
    const float* __restrict__ Wg, const float* __restrict__ Wn,
    unsigned short* __restrict__ Wc) {
  int i = blockIdx.x * blockDim.x + threadIdx.x;     // 0 .. 128*2048-1
  const int half = NEXP * DIM;
  float v = (i < half) ? Wg[i] : Wn[i - half];
  Wc[i] = f32_to_bf16(v);
}

// ---------------------------------------------------------------------------
// Kernel 2: zero the small accumulator block (ws is poisoned with 0xAA)
// fAcc[64] | pAcc[64] | zAcc[1]
// ---------------------------------------------------------------------------
__global__ void moe_zero_acc(float* __restrict__ acc) {
  int t = threadIdx.x;
  if (t < 2 * NEXP + 1) acc[t] = 0.0f;
}

// ---------------------------------------------------------------------------
// Kernel 3: fused dual GEMM  logits = x@Wg^T + bg,  ndot = x@Wn^T
// 256 threads = 8 wave32s. Block tile: 64 rows x 128 cols.
// Wave (mw,nw) = (wave&1, wave>>1) computes 32x32 patch as 2x2 WMMA tiles.
// ---------------------------------------------------------------------------
__global__ __launch_bounds__(256) void moe_gate_gemm(
    const float*          __restrict__ x,      // [B][2048] f32
    const unsigned short* __restrict__ Wc,     // [128][2048] bf16
    const float*          __restrict__ bg,     // [64]
    float*                __restrict__ logits, // [B][64]
    float*                __restrict__ ndot) { // [B][64]
  __shared__ unsigned short As[MB][KB + LDSPAD];     // 64 x 72 bf16  (9.2 KB)
  __shared__ unsigned short Bs[NCOLS][KB + LDSPAD];  // 128 x 72 bf16 (18.4 KB)

  const int tid      = threadIdx.x;
  const int lane     = tid & 31;
  const int wave     = tid >> 5;
  const int rowBlock = blockIdx.x * MB;

  const int mw   = wave & 1;        // 0..1 -> row offset 0/32
  const int nw   = wave >> 1;       // 0..3 -> col offset 0/32/64/96
  const int rowW = mw * 32;
  const int colW = nw * 32;

  v8f acc00 = {}, acc01 = {}, acc10 = {}, acc11 = {};

  // per-lane fragment addressing (ISA 7.12.2 layouts, wave32)
  const int aRow   = rowW + (lane & 15);        // A-tile row (first of two tiles)
  const int aKbase = (lane >> 4) * 8;           // lanes 16-31 hold K=8..15 / 24..31
  const int bCol   = colW + (lane & 15);        // B column
  const int bKoff  = (lane >> 4) * 8;           // lanes 16-31 hold K=8..15 / 24..31

  // cooperative staging indices
  const int ldRowA = tid >> 2;                  // 0..63
  const int ldKA   = (tid & 3) * 16;            // 16 floats per thread
  const int ldColB = tid >> 1;                  // 0..127
  const int ldKB   = (tid & 1) * 32;            // 32 bf16 per thread

  for (int k0 = 0; k0 < DIM; k0 += KB) {
    // stage x tile: 64x64 f32 -> bf16 LDS
    {
      const float4* s4 = reinterpret_cast<const float4*>(
          x + (size_t)(rowBlock + ldRowA) * DIM + k0 + ldKA);
      union { unsigned short u[16]; u32x4 q[2]; } tmp;
#pragma unroll
      for (int i = 0; i < 4; ++i) {
        float4 f = s4[i];
        tmp.u[4 * i + 0] = f32_to_bf16(f.x);
        tmp.u[4 * i + 1] = f32_to_bf16(f.y);
        tmp.u[4 * i + 2] = f32_to_bf16(f.z);
        tmp.u[4 * i + 3] = f32_to_bf16(f.w);
      }
      u32x4* dA = reinterpret_cast<u32x4*>(&As[ldRowA][ldKA]);
      dA[0] = tmp.q[0];
      dA[1] = tmp.q[1];
    }
    // stage weight tile: 128x64 bf16 (from L2-resident Wc)
    {
      const u32x4* sB = reinterpret_cast<const u32x4*>(
          Wc + (size_t)ldColB * DIM + k0 + ldKB);
      u32x4* dB = reinterpret_cast<u32x4*>(&Bs[ldColB][ldKB]);
#pragma unroll
      for (int i = 0; i < 4; ++i) dB[i] = sB[i];
    }
    __syncthreads();

#pragma unroll
    for (int ks = 0; ks < KB; ks += 32) {
      Frag a0, a1, b0, b1;
      a0.q[0] = *reinterpret_cast<const u32x4*>(&As[aRow][ks + aKbase]);
      a0.q[1] = *reinterpret_cast<const u32x4*>(&As[aRow][ks + aKbase + 16]);
      a1.q[0] = *reinterpret_cast<const u32x4*>(&As[aRow + 16][ks + aKbase]);
      a1.q[1] = *reinterpret_cast<const u32x4*>(&As[aRow + 16][ks + aKbase + 16]);
      b0.q[0] = *reinterpret_cast<const u32x4*>(&Bs[bCol][ks + bKoff]);
      b0.q[1] = *reinterpret_cast<const u32x4*>(&Bs[bCol][ks + bKoff + 16]);
      b1.q[0] = *reinterpret_cast<const u32x4*>(&Bs[bCol + 16][ks + bKoff]);
      b1.q[1] = *reinterpret_cast<const u32x4*>(&Bs[bCol + 16][ks + bKoff + 16]);

      acc00 = __builtin_amdgcn_wmma_f32_16x16x32_bf16(false, a0.v, false, b0.v,
                                                      (short)0, acc00, false, false);
      acc01 = __builtin_amdgcn_wmma_f32_16x16x32_bf16(false, a0.v, false, b1.v,
                                                      (short)0, acc01, false, false);
      acc10 = __builtin_amdgcn_wmma_f32_16x16x32_bf16(false, a1.v, false, b0.v,
                                                      (short)0, acc10, false, false);
      acc11 = __builtin_amdgcn_wmma_f32_16x16x32_bf16(false, a1.v, false, b1.v,
                                                      (short)0, acc11, false, false);
    }
    __syncthreads();
  }

  // epilogue: C layout -> row = tileRow + j + 8*(lane>>4), col = lane&15
  const int rHalf = (lane >> 4) * 8;
  const int cLane = lane & 15;
#pragma unroll
  for (int ti = 0; ti < 2; ++ti) {
#pragma unroll
    for (int tj = 0; tj < 2; ++tj) {
      v8f a = (ti == 0) ? (tj == 0 ? acc00 : acc01) : (tj == 0 ? acc10 : acc11);
      int c = colW + tj * 16 + cLane;
      float bias = (c < NEXP) ? bg[c] : 0.0f;
      int rBase = rowBlock + rowW + ti * 16 + rHalf;
#pragma unroll
      for (int e = 0; e < 8; ++e) {
        float v = a[e] + ((c < NEXP) ? bias : 0.0f);
        size_t r = (size_t)(rBase + e);
        if (c < NEXP) logits[r * NEXP + c] = v;
        else          ndot[r * NEXP + (c - NEXP)] = a[e];
      }
    }
  }
}

// ---------------------------------------------------------------------------
// Kernel 4: per-row epilogue — softplus noise, noisy logits, top-8 + softmax,
// full softmax stats for aux losses. One wave32 per row, 8 rows per block.
// ---------------------------------------------------------------------------
__global__ __launch_bounds__(256) void moe_gate_topk(
    const float* __restrict__ logits, const float* __restrict__ ndot,
    const float* __restrict__ noise,
    float* __restrict__ outW, float* __restrict__ outI,
    float* __restrict__ fAcc, float* __restrict__ pAcc, float* __restrict__ zAcc) {
  __shared__ float pSh[NEXP];
  __shared__ float fSh[NEXP];
  __shared__ float zSh;

  const int tid  = threadIdx.x;
  const int lane = tid & 31;
  const int wave = tid >> 5;

  if (tid < NEXP) { pSh[tid] = 0.0f; fSh[tid] = 0.0f; }
  if (tid == 0)   zSh = 0.0f;
  __syncthreads();

  const size_t row = (size_t)blockIdx.x * 8 + wave;
  const size_t base = row * NEXP;

  float l0 = logits[base + lane], l1 = logits[base + lane + 32];
  float n0 = ndot[base + lane],   n1 = ndot[base + lane + 32];
  float z0 = noise[base + lane],  z1 = noise[base + lane + 32];

  float y0 = l0 + z0 * softplus_stable(n0);
  float y1 = l1 + z1 * softplus_stable(n1);

  // ---- full softmax over 64 logits (for p and z_loss) ----
  float m = fmaxf(l0, l1);
#pragma unroll
  for (int off = 16; off > 0; off >>= 1) m = fmaxf(m, __shfl_xor(m, off, 32));
  float e0 = __expf(l0 - m), e1 = __expf(l1 - m);
  float s = e0 + e1;
#pragma unroll
  for (int off = 16; off > 0; off >>= 1) s += __shfl_xor(s, off, 32);
  float inv = 1.0f / s;
  atomicAdd(&pSh[lane], e0 * inv);
  atomicAdd(&pSh[lane + 32], e1 * inv);
  if (lane == 0) {
    float lse = m + __logf(s);
    atomicAdd(&zSh, lse * lse);
  }

  // ---- iterative top-8 over noisy logits (2 candidates per lane) ----
  float c0 = y0, c1 = y1;
  float myv = 0.0f, v0max = 0.0f, s8 = 0.0f;
  int   myi = 0, ix0 = 0;
#pragma unroll
  for (int k = 0; k < TOPK; ++k) {
    float v;
    int ix;
    if (c0 >= c1) { v = c0; ix = lane; } else { v = c1; ix = lane + 32; }
#pragma unroll
    for (int off = 16; off > 0; off >>= 1) {
      float ov = __shfl_xor(v, off, 32);
      int   oi = __shfl_xor(ix, off, 32);
      if (ov > v || (ov == v && oi < ix)) { v = ov; ix = oi; }
    }
    if (k == 0) { v0max = v; ix0 = ix; }
    s8 += __expf(v - v0max);
    if (lane == k) { myv = v; myi = ix; }
    if (ix == lane)           c0 = -3.0e38f;
    else if (ix == lane + 32) c1 = -3.0e38f;
  }

  if (lane == 0) atomicAdd(&fSh[ix0], 1.0f);     // dispatch = argmax one-hot

  float inv8 = 1.0f / s8;
  if (lane < TOPK) {
    outW[row * TOPK + lane] = __expf(myv - v0max) * inv8;
    outI[row * TOPK + lane] = (float)myi;
  }

  __syncthreads();
  if (tid < NEXP) {
    atomicAdd(&pAcc[tid], pSh[tid]);
    atomicAdd(&fAcc[tid], fSh[tid]);
  }
  if (tid == 0) atomicAdd(zAcc, zSh);
}

// ---------------------------------------------------------------------------
// Kernel 5: finalize the two scalar losses
// ---------------------------------------------------------------------------
__global__ void moe_gate_finalize(const float* __restrict__ fAcc,
                                  const float* __restrict__ pAcc,
                                  const float* __restrict__ zAcc,
                                  float* __restrict__ outScalars) {
  int lane = threadIdx.x;   // 32 threads
  float part = fAcc[lane] * pAcc[lane] + fAcc[lane + 32] * pAcc[lane + 32];
#pragma unroll
  for (int off = 16; off > 0; off >>= 1) part += __shfl_xor(part, off, 32);
  if (lane == 0) {
    const float invB = 1.0f / (float)NBATCH;
    outScalars[0] = (float)NEXP * part * invB * invB;  // load_balance_loss
    outScalars[1] = zAcc[0] * invB;                    // z_loss
  }
}

// ---------------------------------------------------------------------------
extern "C" void kernel_launch(void* const* d_in, const int* in_sizes, int n_in,
                              void* d_out, int out_size, void* d_ws, size_t ws_size,
                              hipStream_t stream) {
  const float* x     = (const float*)d_in[0];   // [16384][2048]
  const float* noise = (const float*)d_in[1];   // [16384][64]
  const float* Wg    = (const float*)d_in[2];   // [64][2048]
  const float* bg    = (const float*)d_in[3];   // [64]
  const float* Wn    = (const float*)d_in[4];   // [64][2048]

  // workspace layout
  char* ws = (char*)d_ws;
  unsigned short* Wc = (unsigned short*)ws;                          // 512 KB bf16
  float* logits = (float*)(ws + (size_t)NCOLS * DIM * 2);            // 4 MB
  float* ndot   = (float*)((char*)logits + (size_t)NBATCH * NEXP * 4); // 4 MB
  float* acc    = (float*)((char*)ndot + (size_t)NBATCH * NEXP * 4);   // 129 floats
  float* fAcc = acc;
  float* pAcc = acc + NEXP;
  float* zAcc = acc + 2 * NEXP;

  float* out  = (float*)d_out;
  float* outW = out;                                  // [16384][8] weights
  float* outI = out + (size_t)NBATCH * TOPK;          // [16384][8] indices (as f32)
  float* outS = out + (size_t)2 * NBATCH * TOPK;      // [lb_loss, z_loss]

  moe_convert_weights<<<(NCOLS * DIM) / 256, 256, 0, stream>>>(Wg, Wn, Wc);
  moe_zero_acc<<<1, 256, 0, stream>>>(acc);
  moe_gate_gemm<<<NBATCH / MB, 256, 0, stream>>>(x, Wc, bg, logits, ndot);
  moe_gate_topk<<<NBATCH / 8, 256, 0, stream>>>(logits, ndot, noise,
                                                outW, outI, fAcc, pAcc, zAcc);
  moe_gate_finalize<<<1, 32, 0, stream>>>(fAcc, pAcc, zAcc, outS);
}
